// DeepseekV2DecoderLayer_1571958030520
// MI455X (gfx1250) — matrix operand order repaired and checked by reading, not compiled, not run on hardware
//
#include <hip/hip_runtime.h>
#include <math.h>

// ---------------- types ----------------
typedef __attribute__((ext_vector_type(16))) __bf16 v16bf;
typedef __attribute__((ext_vector_type(8)))  __bf16 v8bf;
typedef __attribute__((ext_vector_type(8)))  float  v8f;
typedef __attribute__((ext_vector_type(4)))  int    v4i;
typedef unsigned short u16;

#define HIDDEN 2048
#define INTER  10944
#define TOK    4096            // B*S = 2*2048
#define EPSV   1e-6f

// LDS B-tile column stride in u16: 32 K-values + 8 pad = 80B = 20 dwords.
// 5*l mod 16 is a permutation -> 16 fragment lanes hit 16 distinct bank groups.
#define LDS_STRIDE 40

// ---------------- async-to-LDS path (gated) ----------------
#if defined(__has_builtin)
#if __has_builtin(__builtin_amdgcn_global_load_async_to_lds_b128) && \
    __has_builtin(__builtin_amdgcn_s_wait_asynccnt)
#define USE_ASYNC 1
#endif
#endif
#ifndef USE_ASYNC
#define USE_ASYNC 0
#endif

#if USE_ASYNC
typedef __attribute__((address_space(1))) v4i* gas1_v4i;
typedef __attribute__((address_space(3))) v4i* las3_v4i;
#define WAIT_ASYNC(n) __builtin_amdgcn_s_wait_asynccnt(n)
#else
#define WAIT_ASYNC(n) do {} while (0)
#endif

// ---------------- helpers ----------------
__device__ __forceinline__ u16 f32_to_bf16(float f) {
    unsigned u = __float_as_uint(f);
    u += 0x7FFFu + ((u >> 16) & 1u);      // round-to-nearest-even
    return (u16)(u >> 16);
}

struct U16x4 { u16 x, y, z, w; };

// ---------------- fp32 -> bf16 conversion (vectorized) ----------------
__global__ void __launch_bounds__(256)
k_cvt_bf16(const float* __restrict__ s, u16* __restrict__ d, long n4) {
    long i = (long)blockIdx.x * 256 + threadIdx.x;
    if (i < n4) {
        float4 f = reinterpret_cast<const float4*>(s)[i];
        U16x4 o;
        o.x = f32_to_bf16(f.x); o.y = f32_to_bf16(f.y);
        o.z = f32_to_bf16(f.z); o.w = f32_to_bf16(f.w);
        reinterpret_cast<U16x4*>(d)[i] = o;
    }
}

// ---------------- RMSNorm: fp32 in -> bf16 out ----------------
__global__ void __launch_bounds__(256)
k_rmsnorm(const float* __restrict__ x, const float* __restrict__ w,
          u16* __restrict__ out) {
    __shared__ float red[256];
    const int row  = blockIdx.x;
    const int base = threadIdx.x * 8;
    const float* xr = x + (size_t)row * HIDDEN;

    float v[8];
    float ss = 0.f;
#pragma unroll
    for (int i = 0; i < 2; ++i) {
        float4 f = *reinterpret_cast<const float4*>(xr + base + i * 4);
        v[i*4+0] = f.x; v[i*4+1] = f.y; v[i*4+2] = f.z; v[i*4+3] = f.w;
        ss += f.x*f.x + f.y*f.y + f.z*f.z + f.w*f.w;
    }
    red[threadIdx.x] = ss;
    __syncthreads();
#pragma unroll
    for (int s = 128; s > 0; s >>= 1) {
        if (threadIdx.x < s) red[threadIdx.x] += red[threadIdx.x + s];
        __syncthreads();
    }
    const float scale = rsqrtf(red[0] * (1.0f / HIDDEN) + EPSV);

    u16* orow = out + (size_t)row * HIDDEN;
#pragma unroll
    for (int i = 0; i < 2; ++i) {
        float4 wf = *reinterpret_cast<const float4*>(w + base + i * 4);
        U16x4 o;
        o.x = f32_to_bf16(v[i*4+0] * scale * wf.x);
        o.y = f32_to_bf16(v[i*4+1] * scale * wf.y);
        o.z = f32_to_bf16(v[i*4+2] * scale * wf.z);
        o.w = f32_to_bf16(v[i*4+3] * scale * wf.w);
        *reinterpret_cast<U16x4*>(orow + base + i * 4) = o;
    }
}

// ---------------- WMMA fragment loaders ----------------
// A: [rows, K] bf16 row-major. ISA 16-bit A layout (16x32):
//   lane m (0-15): VGPR0-3 = K 0..7, VGPR4-7 = K 16..23
//   lane m+16   : VGPR0-3 = K 8..15, VGPR4-7 = K 24..31
__device__ __forceinline__ v16bf load_a_frag(const u16* __restrict__ A, long lda,
                                             int row, int k, int ml, int hl) {
    const u16* p0 = A + (size_t)(row + ml) * lda + k + hl * 8;
    union { v16bf v; v8bf h[2]; } t;
    t.h[0] = *reinterpret_cast<const v8bf*>(p0);
    t.h[1] = *reinterpret_cast<const v8bf*>(p0 + 16);
    return t.v;
}
// B fragment from an LDS-staged tile: lane n = col, K {8h..8h+7, 16+8h..16+8h+7} per lane
__device__ __forceinline__ v16bf lds_b_frag(const u16* lbuf, int jcol, int ml, int hl) {
    const u16* p = lbuf + (jcol + ml) * LDS_STRIDE + hl * 8;
    union { v16bf v; v8bf h[2]; } t;
    t.h[0] = *reinterpret_cast<const v8bf*>(p);
    t.h[1] = *reinterpret_cast<const v8bf*>(p + 16);
    return t.v;
}

// Stage one 64-col x 32-K B tile into LDS. Each thread moves 32 bytes:
// t -> col = t>>1, half = t&1. Works for any 128-thread slice (pass t in 0..127).
__device__ __forceinline__ void stage_b_tile(const u16* __restrict__ B, int K,
                                             int col0, int k, u16* lbuf, int t) {
    const int col  = t >> 1;
    const int half = t & 1;
    const u16* g = B + (size_t)(col0 + col) * K + k + half * 16;
    u16*       l = lbuf + col * LDS_STRIDE + half * 16;
#if USE_ASYNC
    // offset imm applies to both global and LDS addresses (ISA 10.x async ops)
    __builtin_amdgcn_global_load_async_to_lds_b128((gas1_v4i)g, (las3_v4i)l, 0, 0);
    __builtin_amdgcn_global_load_async_to_lds_b128((gas1_v4i)g, (las3_v4i)l, 16, 0);
#else
    v8bf r0 = *reinterpret_cast<const v8bf*>(g);
    v8bf r1 = *reinterpret_cast<const v8bf*>(g + 8);
    *reinterpret_cast<v8bf*>(l)     = r0;
    *reinterpret_cast<v8bf*>(l + 8) = r1;
#endif
}

#define WMMA_BF16(a, b, c) \
    __builtin_amdgcn_wmma_f32_16x16x32_bf16(false, (a), false, (b), (short)0, (c), false, false)

// ---------------- GEMM mainloop (4 waves, wave tile 64x64, WG 256x64) -------
// Computes acc[4][4] for C[M,N] = A[M,K] * B[N,K]^T with B staged in LDS.
#define GEMM_BODY(A_, B_, K_, row0_, col0_, acc_)                                   \
    __shared__ u16 lbs[2][64 * LDS_STRIDE];                                         \
    stage_b_tile(B_, K_, col0_, 0, lbs[0], (int)threadIdx.x);                       \
    int cur = 0;                                                                    \
    for (int k = 0; k < K_; k += 32) {                                              \
        if ((k + 32) < K_) {                                                        \
            stage_b_tile(B_, K_, col0_, k + 32, lbs[cur ^ 1], (int)threadIdx.x);    \
            WAIT_ASYNC(2);                                                          \
        } else {                                                                    \
            WAIT_ASYNC(0);                                                          \
        }                                                                           \
        __syncthreads();                                                            \
        v16bf a[4], b[4];                                                           \
        _Pragma("unroll")                                                           \
        for (int i = 0; i < 4; ++i) a[i] = load_a_frag(A_, K_, row0_ + i * 16, k, ml, hl); \
        _Pragma("unroll")                                                           \
        for (int j = 0; j < 4; ++j) b[j] = lds_b_frag(lbs[cur], j * 16, ml, hl);    \
        _Pragma("unroll")                                                           \
        for (int i = 0; i < 4; ++i)                                                 \
            _Pragma("unroll")                                                       \
            for (int j = 0; j < 4; ++j)                                             \
                acc_[i][j] = WMMA_BF16(a[i], b[j], acc_[i][j]);                     \
        __syncthreads();                                                            \
        cur ^= 1;                                                                   \
    }

// ---------------- GEMM, bf16 out ----------------
__global__ void __launch_bounds__(128)
k_gemm_bf16(const u16* __restrict__ A, const u16* __restrict__ B,
            u16* __restrict__ C, int N, int K) {
    const int lane = threadIdx.x & 31;
    const int wave = threadIdx.x >> 5;
    const int ml = lane & 15, hl = lane >> 4;
    const int row0 = blockIdx.y * 256 + wave * 64;
    const int col0 = blockIdx.x * 64;

    v8f acc[4][4];
#pragma unroll
    for (int i = 0; i < 4; ++i)
#pragma unroll
        for (int j = 0; j < 4; ++j)
#pragma unroll
            for (int r = 0; r < 8; ++r) acc[i][j][r] = 0.f;

    GEMM_BODY(A, B, K, row0, col0, acc)

#pragma unroll
    for (int i = 0; i < 4; ++i)
#pragma unroll
        for (int j = 0; j < 4; ++j) {
            const int ccol = col0 + j * 16 + ml;
#pragma unroll
            for (int r = 0; r < 8; ++r) {
                const int crow = row0 + i * 16 + hl * 8 + r;
                C[(size_t)crow * N + ccol] = f32_to_bf16(acc[i][j][r]);
            }
        }
}

// ---------------- GEMM + residual, fp32 out ----------------
__global__ void __launch_bounds__(128)
k_gemm_res_f32(const u16* __restrict__ A, const u16* __restrict__ B,
               const float* __restrict__ R, float* __restrict__ O, int N, int K) {
    const int lane = threadIdx.x & 31;
    const int wave = threadIdx.x >> 5;
    const int ml = lane & 15, hl = lane >> 4;
    const int row0 = blockIdx.y * 256 + wave * 64;
    const int col0 = blockIdx.x * 64;

    v8f acc[4][4];
#pragma unroll
    for (int i = 0; i < 4; ++i)
#pragma unroll
        for (int j = 0; j < 4; ++j)
#pragma unroll
            for (int r = 0; r < 8; ++r) acc[i][j][r] = 0.f;

    GEMM_BODY(A, B, K, row0, col0, acc)

#pragma unroll
    for (int i = 0; i < 4; ++i)
#pragma unroll
        for (int j = 0; j < 4; ++j) {
            const int ccol = col0 + j * 16 + ml;
#pragma unroll
            for (int r = 0; r < 8; ++r) {
                const int crow = row0 + i * 16 + hl * 8 + r;
                const size_t idx = (size_t)crow * N + ccol;
                O[idx] = acc[i][j][r] + R[idx];
            }
        }
}

// ---------------- fused gate/up GEMM + SiLU, bf16 out ----------------
// 256 threads = 8 waves; wave tile 32x64, dual accumulators (gate & up);
// both B tiles staged in LDS (threads 0..127 -> gate, 128..255 -> up).
__global__ void __launch_bounds__(256)
k_gemm_gateup(const u16* __restrict__ A, const u16* __restrict__ Bg,
              const u16* __restrict__ Bu, u16* __restrict__ GU, int N, int K) {
    const int lane = threadIdx.x & 31;
    const int wave = threadIdx.x >> 5;
    const int ml = lane & 15, hl = lane >> 4;
    const int row0 = blockIdx.y * 256 + wave * 32;
    const int col0 = blockIdx.x * 64;

    __shared__ u16 lg[2][64 * LDS_STRIDE];
    __shared__ u16 lu[2][64 * LDS_STRIDE];

    v8f ag[2][4], au[2][4];
#pragma unroll
    for (int i = 0; i < 2; ++i)
#pragma unroll
        for (int j = 0; j < 4; ++j)
#pragma unroll
            for (int r = 0; r < 8; ++r) { ag[i][j][r] = 0.f; au[i][j][r] = 0.f; }

    const int thalf = (int)threadIdx.x & 127;
    const bool isUp = threadIdx.x >= 128;

    if (!isUp) stage_b_tile(Bg, K, col0, 0, lg[0], thalf);
    else       stage_b_tile(Bu, K, col0, 0, lu[0], thalf);
    int cur = 0;
    for (int k = 0; k < K; k += 32) {
        if ((k + 32) < K) {
            if (!isUp) stage_b_tile(Bg, K, col0, k + 32, lg[cur ^ 1], thalf);
            else       stage_b_tile(Bu, K, col0, k + 32, lu[cur ^ 1], thalf);
            WAIT_ASYNC(2);
        } else {
            WAIT_ASYNC(0);
        }
        __syncthreads();
        v16bf a[2], bg[4], bu[4];
#pragma unroll
        for (int i = 0; i < 2; ++i) a[i] = load_a_frag(A, K, row0 + i * 16, k, ml, hl);
#pragma unroll
        for (int j = 0; j < 4; ++j) {
            bg[j] = lds_b_frag(lg[cur], j * 16, ml, hl);
            bu[j] = lds_b_frag(lu[cur], j * 16, ml, hl);
        }
#pragma unroll
        for (int i = 0; i < 2; ++i)
#pragma unroll
            for (int j = 0; j < 4; ++j) {
                ag[i][j] = WMMA_BF16(a[i], bg[j], ag[i][j]);
                au[i][j] = WMMA_BF16(a[i], bu[j], au[i][j]);
            }
        __syncthreads();
        cur ^= 1;
    }
#pragma unroll
    for (int i = 0; i < 2; ++i)
#pragma unroll
        for (int j = 0; j < 4; ++j) {
            const int ccol = col0 + j * 16 + ml;
#pragma unroll
            for (int r = 0; r < 8; ++r) {
                const int crow = row0 + i * 16 + hl * 8 + r;
                const float g = ag[i][j][r];
                const float u = au[i][j][r];
                const float s = g / (1.f + __expf(-g));   // SiLU
                GU[(size_t)crow * N + ccol] = f32_to_bf16(s * u);
            }
        }
}

// ---------------- workspace layout (bytes, all 256B-aligned) ----------------
static constexpr size_t SZ_HH2 = (size_t)HIDDEN * HIDDEN * 2;  //  8 MB
static constexpr size_t SZ_IH2 = (size_t)INTER  * HIDDEN * 2;  // ~44.8 MB
static constexpr size_t OFF_WQ  = 0;
static constexpr size_t OFF_WO  = OFF_WQ  + SZ_HH2;
static constexpr size_t OFF_WG  = OFF_WO  + SZ_HH2;
static constexpr size_t OFF_WU  = OFF_WG  + SZ_IH2;
static constexpr size_t OFF_WD  = OFF_WU  + SZ_IH2;
static constexpr size_t OFF_HN  = OFF_WD  + SZ_IH2;                       // h_norm bf16
static constexpr size_t OFF_Q   = OFF_HN  + (size_t)TOK * HIDDEN * 2;     // q bf16
static constexpr size_t OFF_HID = OFF_Q   + (size_t)TOK * HIDDEN * 2;     // hidden f32
static constexpr size_t OFF_HP  = OFF_HID + (size_t)TOK * HIDDEN * 4;     // h_post bf16
static constexpr size_t OFF_GU  = OFF_HP  + (size_t)TOK * HIDDEN * 2;     // silu(g)*u bf16

extern "C" void kernel_launch(void* const* d_in, const int* in_sizes, int n_in,
                              void* d_out, int out_size, void* d_ws, size_t ws_size,
                              hipStream_t stream) {
    (void)in_sizes; (void)n_in; (void)out_size; (void)ws_size;
    const float* x      = (const float*)d_in[0];
    // d_in[1] = positions (unused by reference math)
    const float* in_w   = (const float*)d_in[2];
    const float* post_w = (const float*)d_in[3];
    const float* Wq     = (const float*)d_in[4];
    const float* Wo     = (const float*)d_in[5];
    const float* Wg     = (const float*)d_in[6];
    const float* Wu     = (const float*)d_in[7];
    const float* Wd     = (const float*)d_in[8];
    float* out = (float*)d_out;

    char* ws = (char*)d_ws;
    u16* wq_b  = (u16*)(ws + OFF_WQ);
    u16* wo_b  = (u16*)(ws + OFF_WO);
    u16* wg_b  = (u16*)(ws + OFF_WG);
    u16* wu_b  = (u16*)(ws + OFF_WU);
    u16* wd_b  = (u16*)(ws + OFF_WD);
    u16* hn_b  = (u16*)(ws + OFF_HN);
    u16* q_b   = (u16*)(ws + OFF_Q);
    float* hid = (float*)(ws + OFF_HID);
    u16* hp_b  = (u16*)(ws + OFF_HP);
    u16* gu_b  = (u16*)(ws + OFF_GU);

    const long nHH = (long)HIDDEN * HIDDEN;   // 4,194,304
    const long nIH = (long)INTER  * HIDDEN;   // 22,413,312

    // 1) weights -> bf16 (L2-resident afterwards: ~151 MB < 192 MB L2)
    k_cvt_bf16<<<(unsigned)(nHH / 1024), 256, 0, stream>>>(Wq, wq_b, nHH / 4);
    k_cvt_bf16<<<(unsigned)(nHH / 1024), 256, 0, stream>>>(Wo, wo_b, nHH / 4);
    k_cvt_bf16<<<(unsigned)(nIH / 1024), 256, 0, stream>>>(Wg, wg_b, nIH / 4);
    k_cvt_bf16<<<(unsigned)(nIH / 1024), 256, 0, stream>>>(Wu, wu_b, nIH / 4);
    k_cvt_bf16<<<(unsigned)(nIH / 1024), 256, 0, stream>>>(Wd, wd_b, nIH / 4);

    // 2) input RMSNorm
    k_rmsnorm<<<TOK, 256, 0, stream>>>(x, in_w, hn_b);

    // 3) q = h_norm @ Wq^T             [4096 x 2048 x 2048]
    dim3 gHH(HIDDEN / 64, TOK / 256);   // (32, 16)
    k_gemm_bf16<<<gHH, 128, 0, stream>>>(hn_b, wq_b, q_b, HIDDEN, HIDDEN);

    // 4) hidden = x + q @ Wo^T         [4096 x 2048 x 2048]
    k_gemm_res_f32<<<gHH, 128, 0, stream>>>(q_b, wo_b, x, hid, HIDDEN, HIDDEN);

    // 5) post-attention RMSNorm
    k_rmsnorm<<<TOK, 256, 0, stream>>>(hid, post_w, hp_b);

    // 6) gu = silu(h_post @ Wg^T) * (h_post @ Wu^T)   [4096 x 10944 x 2048]
    dim3 gIH(INTER / 64, TOK / 256);    // (171, 16)
    k_gemm_gateup<<<gIH, 256, 0, stream>>>(hp_b, wg_b, wu_b, gu_b, INTER, HIDDEN);

    // 7) out = hidden + gu @ Wd^T      [4096 x 2048 x 10944]
    k_gemm_res_f32<<<gHH, 128, 0, stream>>>(gu_b, wd_b, hid, out, HIDDEN, INTER);
}
